// GraphAttentionLayer_37752762532269
// MI455X (gfx1250) — matrix operand order repaired and checked by reading, not compiled, not run on hardware
//
#include <hip/hip_runtime.h>

typedef __attribute__((ext_vector_type(16))) _Float16 v16h;
typedef __attribute__((ext_vector_type(8)))  float    v8f;

#define N_NODES 8192
#define F_INP   256
#define F_OUT_  128
#define ALPHA_  0.2f
#define NEG_BIG_ (-9.0e15f)
#define NSPLIT  4            // waves per row-block (split along j)
#define JCHUNK  (N_NODES / NSPLIT)

// ---------------- Kernel 1: Wh = h @ W  (WMMA f16, f32 accumulate) -------------
// One wave per 16x16 output tile. 512 m-tiles x 8 n-tiles = 4096 waves.
__global__ void __launch_bounds__(256)
gat_wh_kernel(const float* __restrict__ h, const float* __restrict__ W,
              float* __restrict__ Wh, _Float16* __restrict__ WhT)
{
    int wave = (blockIdx.x * (blockDim.x >> 5)) + (threadIdx.x >> 5);
    int lane = threadIdx.x & 31;
    int half = lane >> 4;          // 0: lanes 0-15, 1: lanes 16-31
    int l    = lane & 15;
    int mt = wave >> 3;            // 0..511
    int nt = wave & 7;             // 0..7
    int m0 = mt * 16, n0 = nt * 16;

    v8f c = {};
    for (int k0 = 0; k0 < F_INP; k0 += 32) {
        // A-matrix 16x32 f16: lane<16 holds K {0..7,16..23}, lane>=16 K {8..15,24..31}
        const float* hrow = h + (size_t)(m0 + l) * F_INP + k0 + half * 8;
        v16h a;
        #pragma unroll
        for (int i = 0; i < 8; ++i) {
            a[i]     = (_Float16)hrow[i];
            a[i + 8] = (_Float16)hrow[16 + i];
        }
        // B-matrix 32x16 f16: lane = N, lane<16 holds K {0..7,16..23}, lane>=16 K {8..15,24..31}
        const float* wcol = W + (size_t)(k0 + half * 8) * F_OUT_ + n0 + l;
        v16h b;
        #pragma unroll
        for (int i = 0; i < 16; ++i) b[i] = (_Float16)wcol[(i + (i >> 3) * 8) * F_OUT_];

        c = __builtin_amdgcn_wmma_f32_16x16x32_f16(false, a, false, b,
                                                   (short)0, c, false, false);
    }
    // C layout: lane<16: VGPR r = (M=r, N=lane); lane>=16: (M=8+r, N=lane-16)
    #pragma unroll
    for (int r = 0; r < 8; ++r) {
        int M = m0 + half * 8 + r;
        int Nc = n0 + l;
        Wh[(size_t)M * F_OUT_ + Nc] = c[r];
        WhT[(size_t)Nc * N_NODES + M] = (_Float16)c[r];
    }
}

// ---------------- Kernel 1b: si = Wh@a1, sj = Wh@a2 ---------------------------
__global__ void __launch_bounds__(256)
gat_scores_kernel(const float* __restrict__ Wh, const float* __restrict__ a,
                  float* __restrict__ si, float* __restrict__ sj)
{
    int row = blockIdx.x * blockDim.x + threadIdx.x;
    if (row >= N_NODES) return;
    const float* w = Wh + (size_t)row * F_OUT_;
    float s1 = 0.f, s2 = 0.f;
    #pragma unroll 4
    for (int n = 0; n < F_OUT_; ++n) {
        float v = w[n];
        s1 += v * a[n];
        s2 += v * a[F_OUT_ + n];
    }
    si[row] = s1;
    sj[row] = s2;
}

// ---------------- Kernel 2: fused masked softmax + attention @ Wh -------------
// 512 blocks x 128 threads. Block = one 16-row block; its 4 waves each stream a
// 2048-column slice of adj (flash-attention split-K), then merge partial
// softmax states + accumulators exactly through LDS.
__global__ void __launch_bounds__(128)
gat_attn_kernel(const int* __restrict__ adj, const float* __restrict__ si,
                const float* __restrict__ sj, const _Float16* __restrict__ WhT,
                float* __restrict__ out)
{
    __shared__ float accbuf[NSPLIT * 8 * 32 * 8];   // [w][t][lane][r]  32 KB
    __shared__ float smax[NSPLIT * 16];
    __shared__ float ssum[NSPLIT * 16];
    __shared__ float sfac[NSPLIT * 16];
    __shared__ float stot[16];

    int w    = threadIdx.x >> 5;       // split index 0..3
    int lane = threadIdx.x & 31;
    int half = lane >> 4;
    int l    = lane & 15;
    int m0   = blockIdx.x * 16;
    int m    = m0 + l;                 // this lane's logit row
    float si_m = si[m];

    float rowmax = -INFINITY;
    float rowsum = 0.0f;
    v8f acc[8];
    #pragma unroll
    for (int t = 0; t < 8; ++t) {
        v8f z = {};
        acc[t] = z;
    }

    int off = half * 16;                // A-layout column offset for this half-wave
    int jlo = w * JCHUNK, jhi = jlo + JCHUNK;
    for (int jb = jlo; jb < jhi; jb += 32) {
        // adj + sj for this lane's 16 columns: jb+off+{0..15}
        const int* arow = adj + (size_t)m * N_NODES + jb + off;
        const float* sjp = sj + jb + off;
        float e[16];
        float tmax = -INFINITY;
        #pragma unroll
        for (int i = 0; i < 8; ++i) {
            int   a0 = arow[i];
            int   a1 = arow[8 + i];
            float s0 = si_m + sjp[i];
            float s1 = si_m + sjp[8 + i];
            s0 = s0 > 0.f ? s0 : ALPHA_ * s0;       // leaky_relu
            s1 = s1 > 0.f ? s1 : ALPHA_ * s1;
            e[i]     = a0 > 0 ? s0 : NEG_BIG_;
            e[i + 8] = a1 > 0 ? s1 : NEG_BIG_;
            tmax = fmaxf(tmax, fmaxf(e[i], e[i + 8]));
        }
        tmax = fmaxf(tmax, __shfl_xor(tmax, 16, 32));    // combine half-waves
        float mnew  = fmaxf(rowmax, tmax);               // always finite
        float scale = __expf(rowmax - mnew);             // first iter: exp(-inf)=0

        v16h pa;
        float psum = 0.f;
        #pragma unroll
        for (int i = 0; i < 16; ++i) {
            float p = __expf(e[i] - mnew);
            psum += p;
            pa[i] = (_Float16)p;
        }
        psum += __shfl_xor(psum, 16, 32);
        rowsum = rowsum * scale + psum;
        rowmax = mnew;

        // rescale accumulators only when some row's max moved (wave-uniform test)
        if (!__all(scale == 1.0f)) {
            float sr[8];
            #pragma unroll
            for (int r = 0; r < 8; ++r) sr[r] = __shfl(scale, half * 8 + r, 32);
            #pragma unroll
            for (int t = 0; t < 8; ++t) {
                #pragma unroll
                for (int r = 0; r < 8; ++r) acc[t][r] *= sr[r];
            }
        }

        // P(16x32) @ Wh(32x128): 8 WMMA, B from transposed f16 Wh (contiguous)
        #pragma unroll
        for (int t = 0; t < 8; ++t) {
            const _Float16* bp = WhT + (size_t)(t * 16 + l) * N_NODES + jb + half * 16;
            v16h b = *(const v16h*)bp;
            acc[t] = __builtin_amdgcn_wmma_f32_16x16x32_f16(false, pa, false, b,
                                                            (short)0, acc[t], false, false);
        }
    }

    // ---- phase 1: publish per-split softmax state --------------------------
    if (lane < 16) {
        smax[w * 16 + l] = rowmax;
        ssum[w * 16 + l] = rowsum;
    }
    __syncthreads();

    // ---- phase 2: merge factors; dump raw accumulators to LDS --------------
    float gmax = -INFINITY;
    #pragma unroll
    for (int w2 = 0; w2 < NSPLIT; ++w2) gmax = fmaxf(gmax, smax[w2 * 16 + l]);
    if (lane < 16) {
        sfac[w * 16 + l] = __expf(smax[w * 16 + l] - gmax);
        if (w == 0) {
            float tot = 0.f;
            #pragma unroll
            for (int w2 = 0; w2 < NSPLIT; ++w2)
                tot += __expf(smax[w2 * 16 + l] - gmax) * ssum[w2 * 16 + l];
            stot[l] = tot;
        }
    }
    #pragma unroll
    for (int t = 0; t < 8; ++t) {
        float* p = accbuf + (size_t)(((w * 8 + t) * 32) + lane) * 8;
        *(float4*)(p)     = make_float4(acc[t][0], acc[t][1], acc[t][2], acc[t][3]);
        *(float4*)(p + 4) = make_float4(acc[t][4], acc[t][5], acc[t][6], acc[t][7]);
    }
    __syncthreads();

    // ---- phase 3: weighted sum over splits, normalize, elu, store ----------
    int t3 = threadIdx.x >> 4;          // output column tile 0..7
    int l3 = threadIdx.x & 15;          // column within tile
    float vals[2][8];
    #pragma unroll
    for (int g = 0; g < 2; ++g)
        #pragma unroll
        for (int r = 0; r < 8; ++r) vals[g][r] = 0.f;

    #pragma unroll
    for (int w2 = 0; w2 < NSPLIT; ++w2) {
        #pragma unroll
        for (int g = 0; g < 2; ++g) {                 // g=0: rows 0-7 (lanes 0-15)
            const float* p = accbuf + (size_t)(((w2 * 8 + t3) * 32) + l3 + g * 16) * 8;
            const float* f = sfac + w2 * 16 + g * 8;  // factors for rows g*8..g*8+7
            #pragma unroll
            for (int r = 0; r < 8; ++r) vals[g][r] += f[r] * p[r];
        }
    }
    #pragma unroll
    for (int g = 0; g < 2; ++g) {
        #pragma unroll
        for (int r = 0; r < 8; ++r) {
            int M = g * 8 + r;
            float v = vals[g][r] / stot[M];
            v = v > 0.f ? v : (__expf(v) - 1.0f);     // elu
            out[(size_t)(m0 + M) * F_OUT_ + t3 * 16 + l3] = v;
        }
    }
}

// ---------------- Launch ------------------------------------------------------
extern "C" void kernel_launch(void* const* d_in, const int* in_sizes, int n_in,
                              void* d_out, int out_size, void* d_ws, size_t ws_size,
                              hipStream_t stream) {
    const float* h   = (const float*)d_in[0];   // 8192 x 256 f32
    const int*   adj = (const int*)d_in[1];     // 8192 x 8192 i32
    const float* W   = (const float*)d_in[2];   // 256 x 128 f32
    const float* a   = (const float*)d_in[3];   // 256 x 1 f32
    float* out = (float*)d_out;                 // 8192 x 128 f32

    char* ws = (char*)d_ws;
    float*    Wh  = (float*)ws;                            // 4 MB
    _Float16* WhT = (_Float16*)(ws + 4194304);             // 2 MB (128 x 8192)
    float*    si  = (float*)(ws + 6291456);                // 32 KB
    float*    sj  = (float*)(ws + 6291456 + 32768);        // 32 KB

    // K1: 512 m-tiles x 8 n-tiles = 4096 waves, 8 waves/block -> 512 blocks
    gat_wh_kernel<<<512, 256, 0, stream>>>(h, W, Wh, WhT);
    // K1b: one thread per row
    gat_scores_kernel<<<N_NODES / 256, 256, 0, stream>>>(Wh, a, si, sj);
    // K2: one block per 16-row block; 4 split-j waves each
    gat_attn_kernel<<<512, 128, 0, stream>>>(adj, si, sj, WhT, out);
}
